// kNNSelfAttention_781684048668
// MI455X (gfx1250) — matrix operand order repaired and checked by reading, not compiled, not run on hardware
//
#include <hip/hip_runtime.h>
#include <hip/hip_bf16.h>

typedef __bf16 bf16;
typedef __attribute__((ext_vector_type(16))) __bf16 v16bf;
typedef __attribute__((ext_vector_type(8)))  float  v8f;

#define TM 128
#define TN 128
#define TK 32
#define LDW (TK + 8)   // padded LDS row (40 bf16 = 80B, 16B aligned)

union Frag16 { v16bf v; uint4 q[2]; };

__device__ inline v8f v8zero() {
  v8f v;
#pragma unroll
  for (int i = 0; i < 8; i++) v[i] = 0.0f;
  return v;
}

__device__ inline unsigned pk2(float a, float b) {
  unsigned short lo = __builtin_bit_cast(unsigned short, (bf16)a);
  unsigned short hi = __builtin_bit_cast(unsigned short, (bf16)b);
  return (unsigned)lo | ((unsigned)hi << 16);
}

__device__ inline uint4 pack8(float4 f0, float4 f1) {
  uint4 q;
  q.x = pk2(f0.x, f0.y); q.y = pk2(f0.z, f0.w);
  q.z = pk2(f1.x, f1.y); q.w = pk2(f1.z, f1.w);
  return q;
}

// Generic shared-memory pointer -> LDS byte offset (LDS aperture keeps the
// workgroup-relative offset in addr[31:0]; HW adds the wave's LDS_BASE).
__device__ inline unsigned lds_off(const void* p) {
  return (unsigned)(uintptr_t)p;
}

// CDNA5 async global->LDS copy (16B per lane), tracked by ASYNCcnt.
__device__ inline void async_b128(unsigned lds_byte_off, const void* gaddr) {
  asm volatile("global_load_async_to_lds_b128 %0, %1, off"
               :: "v"(lds_byte_off), "v"(gaddr) : "memory");
}
__device__ inline void wait_async4() {
  asm volatile("s_wait_asynccnt 0x4" ::: "memory");
}
__device__ inline void wait_async0() {
  asm volatile("s_wait_asynccnt 0x0" ::: "memory");
}

// One BK=32 step: load A/B fragments from LDS per documented 16-bit WMMA
// layouts (lane<16 -> K 0..7 & 16..23 for A and B; lane>=16 shifted),
// then 4x2 = 8 wmma ops for this wave's 64x32 sub-tile.
__device__ inline void wmma_step(const bf16 (*As)[LDW], const bf16 (*Bs)[LDW],
                                 int waveM, int waveN, int half, int lmod,
                                 v8f acc[4][2]) {
  Frag16 a[4], b[2];
#pragma unroll
  for (int fm = 0; fm < 4; fm++) {
    const bf16* p = &As[waveM * 64 + fm * 16 + lmod][0];
    a[fm].q[0] = *(const uint4*)(p + half * 8);
    a[fm].q[1] = *(const uint4*)(p + 16 + half * 8);
  }
#pragma unroll
  for (int fn = 0; fn < 2; fn++) {
    const bf16* p = &Bs[waveN * 32 + fn * 16 + lmod][0];
    b[fn].q[0] = *(const uint4*)(p + half * 8);
    b[fn].q[1] = *(const uint4*)(p + 16 + half * 8);
  }
#pragma unroll
  for (int fm = 0; fm < 4; fm++)
#pragma unroll
    for (int fn = 0; fn < 2; fn++)
      acc[fm][fn] = __builtin_amdgcn_wmma_f32_16x16x32_bf16(
          false, a[fm].v, false, b[fn].v, (short)0, acc[fm][fn], false, false);
}

// ---------------- Kernel 1: xp = x @ W^T  (fp32 in -> bf16 xp and xp^T) ----
__global__ __launch_bounds__(256) void k1_project(const float* __restrict__ x,
                                                  const float* __restrict__ W,
                                                  bf16* __restrict__ xp,
                                                  bf16* __restrict__ xpT) {
  __shared__ __align__(16) bf16 As[TM][LDW];
  __shared__ __align__(16) bf16 Bs[TN][LDW];
  const int K = 512, D = 512, N = 2048;
  const int m0 = blockIdx.y * TM, n0 = blockIdx.x * TN;
  const int t = threadIdx.x;
  const int wave = t >> 5, lane = t & 31, half = lane >> 4, lmod = lane & 15;
  const int waveM = wave >> 2, waveN = wave & 3;
  const int r = t >> 1, hk = t & 1;

  v8f acc[4][2];
#pragma unroll
  for (int i = 0; i < 4; i++)
#pragma unroll
    for (int j = 0; j < 2; j++) acc[i][j] = v8zero();

  for (int k0 = 0; k0 < K; k0 += TK) {
    const float* pa = x + (size_t)(m0 + r) * K + k0 + hk * 16;
    float4 f0 = *(const float4*)(pa + 0), f1 = *(const float4*)(pa + 4);
    float4 f2 = *(const float4*)(pa + 8), f3 = *(const float4*)(pa + 12);
    *(uint4*)&As[r][hk * 16]     = pack8(f0, f1);
    *(uint4*)&As[r][hk * 16 + 8] = pack8(f2, f3);
    const float* pb = W + (size_t)(n0 + r) * K + k0 + hk * 16;
    float4 g0 = *(const float4*)(pb + 0), g1 = *(const float4*)(pb + 4);
    float4 g2 = *(const float4*)(pb + 8), g3 = *(const float4*)(pb + 12);
    *(uint4*)&Bs[r][hk * 16]     = pack8(g0, g1);
    *(uint4*)&Bs[r][hk * 16 + 8] = pack8(g2, g3);
    if (k0 + TK < K) {
      __builtin_prefetch(pa + TK, 0, 0);
      __builtin_prefetch(pb + TK, 0, 0);
    }
    __syncthreads();
    wmma_step(As, Bs, waveM, waveN, half, lmod, acc);
    __syncthreads();
  }

#pragma unroll
  for (int fm = 0; fm < 4; fm++)
#pragma unroll
    for (int fn = 0; fn < 2; fn++)
#pragma unroll
      for (int rr = 0; rr < 8; rr++) {
        int m = m0 + waveM * 64 + fm * 16 + half * 8 + rr;
        int n = n0 + waveN * 32 + fn * 16 + lmod;
        float v = acc[fm][fn][rr];
        xp[(size_t)m * D + n] = (bf16)v;
        int b = m / N, mm = m % N;
        xpT[((size_t)b * D + n) * N + mm] = (bf16)v;
      }
}

// ---------------- Kernel 2: score = xp @ xp^T per batch (bf16 -> fp32) -----
// Double-buffered LDS, all staging via GLOBAL_LOAD_ASYNC_TO_LDS_B128.
__global__ __launch_bounds__(256) void k2_score(const bf16* __restrict__ xp,
                                                float* __restrict__ score) {
  __shared__ __align__(16) bf16 As[2][TM][LDW];
  __shared__ __align__(16) bf16 Bs[2][TN][LDW];
  const int K = 512, N = 2048;
  const int b = blockIdx.z;
  const bf16* xpb = xp + (size_t)b * N * K;
  float* sb = score + (size_t)b * N * N;
  const int m0 = blockIdx.y * TM, n0 = blockIdx.x * TN;
  const int t = threadIdx.x;
  const int wave = t >> 5, lane = t & 31, half = lane >> 4, lmod = lane & 15;
  const int waveM = wave >> 2, waveN = wave & 3;
  const int r = t >> 1, hk = t & 1;

  const bf16* ga_base = xpb + (size_t)(m0 + r) * K + hk * 16;
  const bf16* gb_base = xpb + (size_t)(n0 + r) * K + hk * 16;

  v8f acc[4][2];
#pragma unroll
  for (int i = 0; i < 4; i++)
#pragma unroll
    for (int j = 0; j < 2; j++) acc[i][j] = v8zero();

  // issue one K-tile (4 async ops per thread) into buffer `buf`
  auto issue = [&](int buf, int k0) {
    async_b128(lds_off(&As[buf][r][hk * 16]),     ga_base + k0);
    async_b128(lds_off(&As[buf][r][hk * 16 + 8]), ga_base + k0 + 8);
    async_b128(lds_off(&Bs[buf][r][hk * 16]),     gb_base + k0);
    async_b128(lds_off(&Bs[buf][r][hk * 16 + 8]), gb_base + k0 + 8);
  };

  issue(0, 0);
  int buf = 0;
  for (int k0 = 0; k0 < K; k0 += TK) {
    if (k0 + TK < K) {
      issue(buf ^ 1, k0 + TK);   // prefetch next tile into other buffer
      wait_async4();             // retire current tile's 4 ops (in-order)
    } else {
      wait_async0();
    }
    __syncthreads();
    wmma_step(As[buf], Bs[buf], waveM, waveN, half, lmod, acc);
    __syncthreads();             // protect buffer before it is re-issued
    buf ^= 1;
  }

#pragma unroll
  for (int fm = 0; fm < 4; fm++)
#pragma unroll
    for (int fn = 0; fn < 2; fn++)
#pragma unroll
      for (int rr = 0; rr < 8; rr++) {
        int m = m0 + waveM * 64 + fm * 16 + half * 8 + rr;
        int n = n0 + waveN * 32 + fn * 16 + lmod;
        sb[(size_t)m * N + n] = acc[fm][fn][rr];
      }
}

// -------- Kernel 3: per-row top-32 + multiplicative mask + softmax ---------
__global__ __launch_bounds__(256) void k3_topk_softmax(float* __restrict__ attn) {
  const int N = 2048, TOPK = 32, NT = 256;
  __shared__ float sc[N];
  __shared__ float wk[N];
  __shared__ unsigned char fl[N];
  __shared__ float rv[NT];
  __shared__ int ri[NT];
  const int row = blockIdx.x;          // b*N + n
  const int diag = row & (N - 1);
  float* base = attn + (size_t)row * N;
  const int t = threadIdx.x;

  for (int j = t; j < N; j += NT) { float v = base[j]; sc[j] = v; wk[j] = v; fl[j] = 0; }
  __syncthreads();

  for (int it = 0; it < TOPK; it++) {
    float bv = -INFINITY; int bi = N;
    for (int j = t; j < N; j += NT) { float v = wk[j]; if (v > bv) { bv = v; bi = j; } }
    rv[t] = bv; ri[t] = bi;
    __syncthreads();
    for (int s = NT / 2; s > 0; s >>= 1) {
      if (t < s) {
        float ov = rv[t + s]; int oi = ri[t + s];
        if (ov > rv[t] || (ov == rv[t] && oi < ri[t])) { rv[t] = ov; ri[t] = oi; }
      }
      __syncthreads();
    }
    if (t == 0) { int sel = ri[0]; if (sel < N) { fl[sel] = 1; wk[sel] = -INFINITY; } }
    __syncthreads();
  }
  if (t == 0) fl[diag] = 1;
  __syncthreads();

  // logits = (neighbor ? 1 : -1e19) * score  (faithful multiplicative mask)
  float mx = -INFINITY;
  for (int j = t; j < N; j += NT) {
    float s = sc[j];
    float lg = fl[j] ? s : -1e19f * s;
    wk[j] = lg;
    mx = fmaxf(mx, lg);
  }
  rv[t] = mx; __syncthreads();
  for (int s = NT / 2; s > 0; s >>= 1) { if (t < s) rv[t] = fmaxf(rv[t], rv[t + s]); __syncthreads(); }
  mx = rv[0]; __syncthreads();

  float sm = 0.0f;
  for (int j = t; j < N; j += NT) { float e = __expf(wk[j] - mx); wk[j] = e; sm += e; }
  rv[t] = sm; __syncthreads();
  for (int s = NT / 2; s > 0; s >>= 1) { if (t < s) rv[t] += rv[t + s]; __syncthreads(); }
  float inv = 1.0f / rv[0];
  for (int j = t; j < N; j += NT) base[j] = wk[j] * inv;
}

// ---------------- Kernel 4: h = attention @ xp  (uses xp^T for B frags) ----
// A tile: sync fp32->bf16 conversion path; B tile: async bf16 copy.
__global__ __launch_bounds__(256) void k4_output(const float* __restrict__ attn,
                                                 const bf16* __restrict__ xpT,
                                                 float* __restrict__ h) {
  __shared__ __align__(16) bf16 As[TM][LDW];
  __shared__ __align__(16) bf16 Bs[TN][LDW];
  const int N = 2048, D = 512;
  const int b = blockIdx.z;
  const float* ab = attn + (size_t)b * N * N;
  const bf16* xtb = xpT + (size_t)b * D * N;
  float* hb = h + (size_t)b * N * D;
  const int m0 = blockIdx.y * TM, n0 = blockIdx.x * TN;
  const int t = threadIdx.x;
  const int wave = t >> 5, lane = t & 31, half = lane >> 4, lmod = lane & 15;
  const int waveM = wave >> 2, waveN = wave & 3;
  const int r = t >> 1, hk = t & 1;

  const bf16* gb_base = xtb + (size_t)(n0 + r) * N + hk * 16;

  v8f acc[4][2];
#pragma unroll
  for (int i = 0; i < 4; i++)
#pragma unroll
    for (int j = 0; j < 2; j++) acc[i][j] = v8zero();

  for (int k0 = 0; k0 < N; k0 += TK) {
    // B: async bf16 copy straight into LDS (prior barrier protects reuse)
    async_b128(lds_off(&Bs[r][hk * 16]),     gb_base + k0);
    async_b128(lds_off(&Bs[r][hk * 16 + 8]), gb_base + k0 + 8);
    // A: fp32 attention -> bf16 via VGPRs
    const float* pa = ab + (size_t)(m0 + r) * N + k0 + hk * 16;
    float4 f0 = *(const float4*)(pa + 0), f1 = *(const float4*)(pa + 4);
    float4 f2 = *(const float4*)(pa + 8), f3 = *(const float4*)(pa + 12);
    *(uint4*)&As[r][hk * 16]     = pack8(f0, f1);
    *(uint4*)&As[r][hk * 16 + 8] = pack8(f2, f3);
    if (k0 + TK < N) __builtin_prefetch(pa + TK, 0, 0);
    wait_async0();               // __syncthreads does not cover ASYNCcnt
    __syncthreads();
    wmma_step(As, Bs, waveM, waveN, half, lmod, acc);
    __syncthreads();
  }

#pragma unroll
  for (int fm = 0; fm < 4; fm++)
#pragma unroll
    for (int fn = 0; fn < 2; fn++)
#pragma unroll
      for (int rr = 0; rr < 8; rr++) {
        int m = m0 + waveM * 64 + fm * 16 + half * 8 + rr;
        int n = n0 + waveN * 32 + fn * 16 + lmod;
        hb[(size_t)m * D + n] = acc[fm][fn][rr];
      }
}

extern "C" void kernel_launch(void* const* d_in, const int* in_sizes, int n_in,
                              void* d_out, int out_size, void* d_ws, size_t ws_size,
                              hipStream_t stream) {
  (void)in_sizes; (void)n_in; (void)out_size; (void)ws_size;
  const int B = 8, N = 2048, D = 512;
  const float* x = (const float*)d_in[0];
  const float* W = (const float*)d_in[1];

  float* h    = (float*)d_out;                         // [B,N,D]
  float* attn = h + (size_t)B * N * D;                 // [B,N,N] (score -> attention in place)

  bf16* xp  = (bf16*)d_ws;                             // [B,N,D] bf16
  bf16* xpT = xp + (size_t)B * N * D;                  // [B,D,N] bf16

  k1_project<<<dim3(D / TN, (B * N) / TM), 256, 0, stream>>>(x, W, xp, xpT);
  k2_score<<<dim3(N / TN, N / TM, B), 256, 0, stream>>>(xp, attn);
  k3_topk_softmax<<<dim3(B * N), 256, 0, stream>>>(attn);
  k4_output<<<dim3(D / TN, N / TM, B), 256, 0, stream>>>(attn, xpT, h);
}
